// GeometricAttention_51041391346137
// MI455X (gfx1250) — compile-verified, matrix operand order checked by reading, not run on hardware
//
#include <hip/hip_runtime.h>
#include <hip/hip_bf16.h>
#include <math.h>

#define NN 8192
#define EE 128
#define HH 4
#define DHD 32
#define KB 64   // keys per attention inner iteration

typedef __bf16 bf16;
typedef bf16  v8bf  __attribute__((ext_vector_type(8)));
typedef bf16  v16bf __attribute__((ext_vector_type(16)));
typedef float v8f   __attribute__((ext_vector_type(8)));

// ---- fragment helpers -------------------------------------------------------
// 16-bit A-matrix 16x32 layout (ISA 7.12.2): lane l holds row, with
// K = (l/16)*8 .. +7 in VGPR0..3 and K = 16+(l/16)*8 .. +7 in VGPR4..7.
// B 32x16 mirrors this with N in place of M. Both reduce to: two contiguous
// 8xbf16 (16B) loads from a 32-element contiguous run per lane.
__device__ __forceinline__ v16bf cat8(v8bf lo, v8bf hi) {
    return __builtin_shufflevector(lo, hi, 0,1,2,3,4,5,6,7,8,9,10,11,12,13,14,15);
}
template <typename P>
__device__ __forceinline__ v16bf load_frag32(const P* base, int lh) {
    v8bf lo = *reinterpret_cast<const v8bf*>(base + lh * 8);
    v8bf hi = *reinterpret_cast<const v8bf*>(base + 16 + lh * 8);
    return cat8(lo, hi);
}
__device__ __forceinline__ v8f wmma_bf16(v16bf a, v16bf b, v8f c) {
    return __builtin_amdgcn_wmma_f32_16x16x32_bf16(false, a, false, b, (short)0, c,
                                                   false, false);
}
// Raw v_exp_f32: single TRANS op. Inputs here are always <= 0; flush-to-zero
// underflow is the correct limit for both the Gaussian bias and softmax tails.
__device__ __forceinline__ float fexp2(float x) { return __builtin_amdgcn_exp2f(x); }

// ---- 1) fp32 -> bf16 convert ------------------------------------------------
__global__ void cvt_f32_bf16(const float* __restrict__ src, bf16* __restrict__ dst, int n) {
    int i = blockIdx.x * blockDim.x + threadIdx.x;
    if (i < n) dst[i] = (bf16)src[i];
}

// ---- 2) projection x @ W^T + b, one matrix per instantiation ----------------
// MAT: 0 = Q (row-major, pre-scaled), 1 = K (row-major), 2 = V (transposed)
template <int MAT>
__global__ void proj_mat(const bf16* __restrict__ fbf, const bf16* __restrict__ Wb,
                         const float* __restrict__ bias, bf16* __restrict__ dst,
                         float scale) {
    const int lane = threadIdx.x & 31;
    const int ll = lane & 15, lh = lane >> 4;
    const int wid = blockIdx.x * (blockDim.x >> 5) + (threadIdx.x >> 5);
    const int rowbase = wid * 16;

    v16bf afrag[4];
#pragma unroll
    for (int kc = 0; kc < 4; ++kc)
        afrag[kc] = load_frag32(fbf + (size_t)(rowbase + ll) * EE + kc * 32, lh);

#pragma unroll
    for (int ct = 0; ct < 8; ++ct) {
        v8f acc = {};
#pragma unroll
        for (int kc = 0; kc < 4; ++kc) {
            // B[k][e] = W[e][k]: column e == W row e, contiguous in k.
            v16bf bfrag = load_frag32(Wb + (size_t)(ct * 16 + ll) * EE + kc * 32, lh);
            acc = wmma_bf16(afrag[kc], bfrag, acc);
        }
        const int col = ct * 16 + ll;
        const float bb = bias[col];
        const int head = col >> 5, dh = col & 31;
#pragma unroll
        for (int v = 0; v < 8; ++v) {
            const int row = rowbase + v + 8 * lh;     // C layout: M = v + 8*(lane/16)
            float val = acc[v] + bb;
            if (MAT == 0) val *= scale;               // fold DH^-0.5 into Q
            if (MAT == 2)                             // V stored transposed [H][DH][N]
                dst[(size_t)head * DHD * NN + (size_t)dh * NN + row] = (bf16)val;
            else                                      // Q/K row-major [H][N][DH]
                dst[(size_t)head * NN * DHD + (size_t)row * DHD + dh] = (bf16)val;
        }
    }
}

// ---- 3) fused flash attention with geometric bias ---------------------------
// __launch_bounds__(256, 1): allow the allocator to use as many VGPRs as it
// needs (wave32 can address up to 1024) instead of spilling qfrag/pointers to
// scratch inside the 128-iteration hot loop. ILP (8 rows x 4 sub-tiles, WMMA
// co-execution) covers the reduced occupancy.
__global__ __launch_bounds__(256, 1) void geo_attn(
        const bf16* __restrict__ Q, const bf16* __restrict__ K,
        const bf16* __restrict__ Vt,
        const float* __restrict__ pos,
        const float* __restrict__ gamma_p, const float* __restrict__ sigma_p,
        bf16* __restrict__ Obf) {
    __shared__ bf16 lds[8 * 16 * KB];                 // 2KB P-staging slot per wave
    const int lane = threadIdx.x & 31;
    const int ll = lane & 15, lh = lane >> 4;
    const int wslot = threadIdx.x >> 5;
    const int wid = blockIdx.x * (blockDim.x >> 5) + wslot;
    const int tilesPerHead = NN / 16;
    const int head = wid / tilesPerHead;
    const int rowbase = (wid % tilesPerHead) * 16;

    const bf16* Qh = Q + (size_t)head * NN * DHD;
    const bf16* Kh = K + (size_t)head * NN * DHD;
    const bf16* Vh = Vt + (size_t)head * DHD * NN;

    const v16bf qfrag = load_frag32(Qh + (size_t)(rowbase + ll) * DHD, lh);

    float pix[8], piy[8], piz[8];
#pragma unroll
    for (int v = 0; v < 8; ++v) {
        const int r = rowbase + v + 8 * lh;
        pix[v] = pos[3 * r + 0]; piy[v] = pos[3 * r + 1]; piz[v] = pos[3 * r + 2];
    }

    const float L2E = 1.4426950408889634f;
    const float gamma = gamma_p[0], sigma = sigma_p[0];
    const float cb = -L2E / (2.0f * sigma * sigma);   // bias = gamma * exp2(d2*cb)

    float m[8], lsum[8];
    v8f o0 = {}, o1 = {};
#pragma unroll
    for (int v = 0; v < 8; ++v) { m[v] = -1e30f; lsum[v] = 0.0f; }

    bf16* myP = lds + wslot * 16 * KB;

    // Incrementing base pointers: K rows advance 64 rows (4KB), V columns
    // advance 64 keys (128B), positions advance 64*3 floats per iteration.
    const bf16* kI  = Kh + (size_t)ll * DHD;          // + t*16*DHD per sub-tile
    const bf16* vI0 = Vh + (size_t)ll * NN;
    const bf16* vI1 = Vh + (size_t)(16 + ll) * NN;
    const float* pj = pos + 3 * ll;
    const bf16* kPf = Kh + (size_t)(KB + 2 * lane) * DHD;   // prefetch bases
    const bf16* vPf = Vh + (size_t)lane * NN + KB;

    for (int kb = 0; kb < NN / KB; ++kb) {
        // prefetch next block into WGP$ (global_prefetch_b8)
        if (kb + 1 < NN / KB) {
            __builtin_prefetch(kPf, 0, 3);
            __builtin_prefetch(vPf, 0, 3);
        }

        // ---- issue K-fragment + position loads together (one clause), so the
        //      position-load latency hides under the 4 QK^T WMMAs ----
        v16bf kf[4];
#pragma unroll
        for (int t = 0; t < 4; ++t) kf[t] = load_frag32(kI + t * 16 * DHD, lh);
        float jx[4], jy[4], jz[4];
#pragma unroll
        for (int t = 0; t < 4; ++t) {
            jx[t] = pj[48 * t + 0]; jy[t] = pj[48 * t + 1]; jz[t] = pj[48 * t + 2];
        }
        v8f s[4];
#pragma unroll
        for (int t = 0; t < 4; ++t) {
            v8f zero = {};
            s[t] = wmma_bf16(qfrag, kf[t], zero);
        }

        // ---- geometric bias (col fixed per lane per 16-wide sub-tile) ----
#pragma unroll
        for (int t = 0; t < 4; ++t)
#pragma unroll
            for (int v = 0; v < 8; ++v) {
                const float dx = pix[v] - jx[t], dy = piy[v] - jy[t], dz = piz[v] - jz[t];
                s[t][v] += gamma * fexp2((dx * dx + dy * dy + dz * dz) * cb);
            }

        // ---- online softmax; stage P to LDS immediately (short p lifetime) ----
#pragma unroll
        for (int v = 0; v < 8; ++v) {
            float t0 = fmaxf(fmaxf(s[0][v], s[1][v]), fmaxf(s[2][v], s[3][v]));
#pragma unroll
            for (int off = 1; off < 16; off <<= 1) t0 = fmaxf(t0, __shfl_xor(t0, off, 32));
            const float mn = fmaxf(m[v], t0);
            const float f = fexp2((m[v] - mn) * L2E);
            m[v] = mn;
            lsum[v] *= f; o0[v] *= f; o1[v] *= f;
            const int r = (v + 8 * lh) * KB + ll;
            float ps = 0.0f;
#pragma unroll
            for (int t = 0; t < 4; ++t) {
                const float pt = fexp2((s[t][v] - mn) * L2E);
                ps += pt;
                myP[r + 16 * t] = (bf16)pt;
            }
#pragma unroll
            for (int off = 1; off < 16; off <<= 1) ps += __shfl_xor(ps, off, 32);
            lsum[v] += ps;
        }

        // ---- LDS -> two A-frags; O += P * V (4 WMMAs) ----
        const v16bf pf0 = load_frag32(myP + ll * KB, lh);        // keys 0..31
        const v16bf pf1 = load_frag32(myP + ll * KB + 32, lh);   // keys 32..63
        const v16bf vf00 = load_frag32(vI0, lh);
        const v16bf vf01 = load_frag32(vI0 + 32, lh);
        const v16bf vf10 = load_frag32(vI1, lh);
        const v16bf vf11 = load_frag32(vI1 + 32, lh);
        o0 = wmma_bf16(pf0, vf00, o0);
        o0 = wmma_bf16(pf1, vf01, o0);
        o1 = wmma_bf16(pf0, vf10, o1);
        o1 = wmma_bf16(pf1, vf11, o1);

        kI += (size_t)KB * DHD;
        vI0 += KB; vI1 += KB;
        pj += 3 * KB;
        kPf += (size_t)KB * DHD;
        vPf += KB;
    }

#pragma unroll
    for (int v = 0; v < 8; ++v) {
        const int row = rowbase + v + 8 * lh;
        const float inv = 1.0f / lsum[v];
        Obf[(size_t)row * EE + head * DHD + ll] = (bf16)(o0[v] * inv);
        Obf[(size_t)row * EE + head * DHD + 16 + ll] = (bf16)(o1[v] * inv);
    }
}

// ---- 4) output projection: O @ Wo^T + bo -> fp32 ----------------------------
__global__ void out_proj(const bf16* __restrict__ Obf, const bf16* __restrict__ Wob,
                         const float* __restrict__ bo, float* __restrict__ out) {
    const int lane = threadIdx.x & 31;
    const int ll = lane & 15, lh = lane >> 4;
    const int wid = blockIdx.x * (blockDim.x >> 5) + (threadIdx.x >> 5);
    const int rowbase = wid * 16;

    v16bf afrag[4];
#pragma unroll
    for (int kc = 0; kc < 4; ++kc)
        afrag[kc] = load_frag32(Obf + (size_t)(rowbase + ll) * EE + kc * 32, lh);

#pragma unroll
    for (int ct = 0; ct < 8; ++ct) {
        v8f acc = {};
#pragma unroll
        for (int kc = 0; kc < 4; ++kc) {
            v16bf bfrag = load_frag32(Wob + (size_t)(ct * 16 + ll) * EE + kc * 32, lh);
            acc = wmma_bf16(afrag[kc], bfrag, acc);
        }
        const int col = ct * 16 + ll;
        const float bb = bo[col];
#pragma unroll
        for (int v = 0; v < 8; ++v) {
            const int row = rowbase + v + 8 * lh;
            out[(size_t)row * EE + col] = acc[v] + bb;
        }
    }
}

// ---- host launcher ----------------------------------------------------------
extern "C" void kernel_launch(void* const* d_in, const int* in_sizes, int n_in,
                              void* d_out, int out_size, void* d_ws, size_t ws_size,
                              hipStream_t stream) {
    const float* features = (const float*)d_in[0];
    const float* positions = (const float*)d_in[1];
    const float* Wq = (const float*)d_in[2];  const float* bq = (const float*)d_in[3];
    const float* Wk = (const float*)d_in[4];  const float* bk = (const float*)d_in[5];
    const float* Wv = (const float*)d_in[6];  const float* bv = (const float*)d_in[7];
    const float* Wo = (const float*)d_in[8];  const float* bo = (const float*)d_in[9];
    const float* gamma = (const float*)d_in[10];
    const float* sigma = (const float*)d_in[11];
    float* out = (float*)d_out;

    char* ws = (char*)d_ws;
    size_t off = 0;
    auto carve = [&](size_t bytes) -> void* {
        void* p = ws + off;
        off += (bytes + 255) & ~(size_t)255;
        return p;
    };
    bf16* fbf = (bf16*)carve((size_t)NN * EE * 2);
    bf16* Wqb = (bf16*)carve((size_t)EE * EE * 2);
    bf16* Wkb = (bf16*)carve((size_t)EE * EE * 2);
    bf16* Wvb = (bf16*)carve((size_t)EE * EE * 2);
    bf16* Wob = (bf16*)carve((size_t)EE * EE * 2);
    bf16* Qb  = (bf16*)carve((size_t)HH * NN * DHD * 2);
    bf16* Kb  = (bf16*)carve((size_t)HH * NN * DHD * 2);
    bf16* Vtb = (bf16*)carve((size_t)HH * DHD * NN * 2);
    bf16* Obf = (bf16*)carve((size_t)NN * EE * 2);

    cvt_f32_bf16<<<(NN * EE + 255) / 256, 256, 0, stream>>>(features, fbf, NN * EE);
    cvt_f32_bf16<<<(EE * EE + 255) / 256, 256, 0, stream>>>(Wq, Wqb, EE * EE);
    cvt_f32_bf16<<<(EE * EE + 255) / 256, 256, 0, stream>>>(Wk, Wkb, EE * EE);
    cvt_f32_bf16<<<(EE * EE + 255) / 256, 256, 0, stream>>>(Wv, Wvb, EE * EE);
    cvt_f32_bf16<<<(EE * EE + 255) / 256, 256, 0, stream>>>(Wo, Wob, EE * EE);

    const float scale = 0.17677669529663687f;  // DH^-0.5, DH = 32
    // 512 row-tiles per matrix, 8 waves/block
    proj_mat<0><<<64, 256, 0, stream>>>(fbf, Wqb, bq, Qb, scale);
    proj_mat<1><<<64, 256, 0, stream>>>(fbf, Wkb, bk, Kb, scale);
    proj_mat<2><<<64, 256, 0, stream>>>(fbf, Wvb, bv, Vtb, scale);
    // 4 heads * 512 query tiles = 2048 wave-tiles
    geo_attn<<<256, 256, 0, stream>>>(Qb, Kb, Vtb, positions, gamma, sigma, Obf);
    // 512 row-tiles
    out_proj<<<64, 256, 0, stream>>>(Obf, Wob, bo, out);
}
